// ANN_47253230190856
// MI455X (gfx1250) — compile-verified, matrix operand order checked by reading, not run on hardware
//
#include <hip/hip_runtime.h>
#include <hip/hip_bf16.h>
#include <math.h>

#define BB 4
#define HH 32
#define SS 8192
#define DD 128
#define RANK_ 32
#define KK 512
#define LOCALK 64
#define BH (BB*HH)
#define CAND 1024

typedef float v2f  __attribute__((ext_vector_type(2)));
typedef float v8f  __attribute__((ext_vector_type(8)));
typedef _Float16 v16h __attribute__((ext_vector_type(16)));

#if __has_builtin(__builtin_amdgcn_wmma_f32_16x16x4_f32)
#define USE_WMMA_F32 1
#else
#define USE_WMMA_F32 0
#endif

// ---------------- K0: per-(b,h) top-32 of |q|, store comp indices + q_proj ----
__global__ void ann_prep(const float* __restrict__ query,
                         float* __restrict__ qproj, int* __restrict__ comp) {
  int bh = blockIdx.x;
  int d  = threadIdx.x; // 0..127
  __shared__ float qs[DD];
  __shared__ unsigned long long kv[DD];
  __shared__ unsigned long long rr[DD];
  float q = query[(size_t)bh*DD + d];
  qs[d] = q;
  float a = fabsf(q); // >= 0, so raw bits are order-preserving as uint
  kv[d] = ((unsigned long long)__float_as_uint(a) << 32) | (unsigned)(DD-1-d);
  __syncthreads();
  for (int it = 0; it < RANK_; ++it) {
    rr[d] = kv[d];
    __syncthreads();
    for (int off = DD/2; off > 0; off >>= 1) {
      if (d < off) rr[d] = rr[d] > rr[d+off] ? rr[d] : rr[d+off];
      __syncthreads();
    }
    unsigned long long win = rr[0];
    int widx = (DD-1) - (int)(win & 0xffffffffull);
    if (d == widx) {
      comp [bh*RANK_ + it] = d;
      qproj[bh*RANK_ + it] = qs[d];
      kv[d] = 0ull; // remove from further consideration
    }
    __syncthreads();
  }
}

// ---------------- K1: rank-32 scores for all S rows via WMMA ------------------
__global__ void ann_scores(const float* __restrict__ key_t,
                           const float* __restrict__ logmask,
                           const float* __restrict__ qproj,
                           const int* __restrict__ comp,
                           float* __restrict__ scores) {
  int bh    = blockIdx.x;
  int chunk = blockIdx.y;           // 8 chunks of 1024 rows
  __shared__ float sq[RANK_];
  __shared__ int   scp[RANK_];
  int tid = threadIdx.x;            // 256 threads = 8 waves
  if (tid < RANK_) { sq[tid] = qproj[bh*RANK_+tid]; scp[tid] = comp[bh*RANK_+tid]; }
  __syncthreads();
  int wave = tid >> 5;
  int lane = tid & 31;
  const float scale = 0.08838834764831845f; // 128^-0.5
  const float* kbase = key_t + (size_t)bh * SS * DD;

  for (int it = 0; it < 8; ++it) {
    int row_base = chunk*1024 + it*128 + wave*16;
    int myrow = row_base + (lane & 15);            // A layout: lane%16 == M
    const float* krow = kbase + (size_t)myrow * DD;
    v8f acc = {};
#if USE_WMMA_F32
    // A 16x4 f32 layout: lanes 0-15 hold K={c,c+1}; lanes 16-31 hold K={c+2,c+3}
    int ksel = (lane < 16) ? 0 : 2;
#pragma unroll
    for (int c = 0; c < 8; ++c) {
      int k0 = c*4 + ksel;
      v2f a, b;
      a[0] = krow[scp[k0]];
      a[1] = krow[scp[k0+1]];
      b[0] = sq[k0];          // q broadcast across all 16 N columns
      b[1] = sq[k0+1];
      acc = __builtin_amdgcn_wmma_f32_16x16x4_f32(false, a, false, b,
                                                  (short)0, acc, false, false);
    }
#else
    // fallback: single 16x16x32 f16 WMMA covering all 32 rank components
    v16h a, b;
    int hi = (lane < 16) ? 0 : 8;
#pragma unroll
    for (int e = 0; e < 16; ++e) {
      int kk2 = ((e < 8) ? e : e + 8) + hi;   // K mapping of 16-bit A layout
      a[e] = (_Float16)krow[scp[kk2]];
      b[e] = (_Float16)sq[kk2];
    }
    acc = __builtin_amdgcn_wmma_f32_16x16x32_f16(false, a, false, b,
                                                 (short)0, acc, false, false);
#endif
    // D layout: VGPR r -> M=r (lanes 0-15) / M=8+r (lanes 16-31); all N equal
    if (lane == 0 || lane == 16) {
      int r0 = row_base + ((lane == 0) ? 0 : 8);
#pragma unroll
      for (int r = 0; r < 8; ++r) {
        int s = r0 + r;
        scores[(size_t)bh*SS + s] = acc[r]*scale + logmask[(size_t)bh*SS + s];
      }
    }
  }
}

// ---------------- K2: partial value sums (streaming, deterministic) ----------
__global__ void ann_vmean_partial(const float* __restrict__ value,
                                  float* __restrict__ vpartial) {
  int bh    = blockIdx.x;
  int chunk = blockIdx.y; // 8 chunks of 1024 rows
  int tid   = threadIdx.x; // 256
  int d = tid & (DD-1);
  int g = tid >> 7;        // 0..1
  const float* vbase = value + (size_t)bh*SS*DD;
  float acc = 0.f;
  for (int r = chunk*1024 + g; r < chunk*1024 + 1024; r += 2) {
    __builtin_prefetch(vbase + (size_t)(r+16)*DD + d, 0, 1);
    acc += vbase[(size_t)r*DD + d];
  }
  __shared__ float p[256];
  p[tid] = acc;
  __syncthreads();
  if (tid < DD) vpartial[((size_t)bh*8 + chunk)*DD + d] = p[tid] + p[tid + DD];
}

// ---------------- K3: logsumexp + top-512 via radix select -------------------
__global__ __launch_bounds__(512) void ann_topk(
    const float* __restrict__ scores, const float* __restrict__ logmask,
    int* __restrict__ indices, float* __restrict__ lm_out) {
  int bh  = blockIdx.x;
  int tid = threadIdx.x; // 512
  extern __shared__ char smem[];
  unsigned* ub              = (unsigned*)smem;                       // 8192 u32
  unsigned long long* cand  = (unsigned long long*)(smem + 32768);   // 1024 u64
  float* red                = (float*)(smem + 32768 + 8192);         // 512 f32
  unsigned* hist            = (unsigned*)(smem + 32768 + 8192 + 2048); // 256
  unsigned* ctl             = hist + 256; // [0]=prefix [1]=mask [2]=need [3]=cnt
  const float* sc = scores  + (size_t)bh*SS;
  const float* lm = logmask + (size_t)bh*SS;

  // ---- ordinals (with local-64 override) + max ----
  float lmax = -INFINITY;
  for (int s = tid; s < SS; s += 512) {
    float v = sc[s];
    lmax = fmaxf(lmax, v);
    float eff = v;
    // causal_index = S-1-s < LOCAL_K and position valid
    if (s >= SS - LOCALK && expf(lm[s]) != 0.0f) eff = 3.402823466e+38f;
    unsigned u = __float_as_uint(eff);
    u = (u & 0x80000000u) ? ~u : (u | 0x80000000u); // order-preserving map
    ub[s] = u;
  }
  red[tid] = lmax; __syncthreads();
  for (int off = 256; off > 0; off >>= 1) {
    if (tid < off) red[tid] = fmaxf(red[tid], red[tid+off]);
    __syncthreads();
  }
  float m = red[0];
  __syncthreads();
  float lsum = 0.f;
  for (int s = tid; s < SS; s += 512) lsum += expf(sc[s] - m);
  red[tid] = lsum; __syncthreads();
  for (int off = 256; off > 0; off >>= 1) {
    if (tid < off) red[tid] += red[tid+off];
    __syncthreads();
  }
  float norm = m + logf(red[0]);
  if (tid == 0) { ctl[0] = 0u; ctl[1] = 0u; ctl[2] = KK; ctl[3] = 0u; }
  __syncthreads();

  // ---- radix select: find exact 32-bit ordinal threshold of 512th largest ----
#pragma unroll
  for (int pass = 0; pass < 4; ++pass) {
    int shift = 24 - pass*8;
    if (tid < 256) hist[tid] = 0u;
    __syncthreads();
    unsigned prefix = ctl[0], mask = ctl[1];
    for (int s = tid; s < SS; s += 512) {
      unsigned u = ub[s];
      if ((u & mask) == prefix) atomicAdd(&hist[(u >> shift) & 255u], 1u);
    }
    __syncthreads();
    if (tid == 0) {
      unsigned need = ctl[2];
      unsigned cum = 0; int bsel = 0;
      for (int b = 255; b >= 0; --b) {
        unsigned h = hist[b];
        if (cum + h >= need) { bsel = b; break; }
        cum += h;
      }
      ctl[2] = need - cum;                       // remaining inside chosen bin
      ctl[0] = prefix | ((unsigned)bsel << shift);
      ctl[1] = mask | (0xFFu << shift);
    }
    __syncthreads();
  }
  unsigned T = ctl[0]; // exact ordinal of the 512th largest value

  // ---- compact candidates (> T definite, == T ties; keys unique via index) --
  for (int s = tid; s < SS; s += 512) {
    unsigned u = ub[s];
    if (u >= T) {
      unsigned slot = atomicAdd(&ctl[3], 1u);
      if (slot < CAND)
        cand[slot] = ((unsigned long long)u << 32) | (unsigned)(SS-1-s);
    }
  }
  __syncthreads();
  unsigned ncand = ctl[3] < CAND ? ctl[3] : CAND;
  for (int i = tid; i < CAND; i += 512)
    if ((unsigned)i >= ncand) cand[i] = 0ull;
  __syncthreads();

  // ---- bitonic sort 1024 candidates, descending ----
  for (int k = 2; k <= CAND; k <<= 1) {
    for (int j = k >> 1; j > 0; j >>= 1) {
      for (int i = tid; i < CAND; i += 512) {
        int ixj = i ^ j;
        if (ixj > i) {
          unsigned long long a = cand[i], b = cand[ixj];
          bool desc = ((i & k) == 0);
          if (desc ? (a < b) : (a > b)) { cand[i] = b; cand[ixj] = a; }
        }
      }
      __syncthreads();
    }
  }

  // ---- first 512 sorted candidates == reference top-k order ----
  int s = SS - 1 - (int)(cand[tid] & 0xffffffffull);
  indices[bh*KK + tid] = s;
  float p = expf(sc[s] - norm);
  red[tid] = p; __syncthreads();
  for (int off = 256; off > 0; off >>= 1) {
    if (tid < off) red[tid] += red[tid+off];
    __syncthreads();
  }
  if (tid == 0) lm_out[(size_t)bh*(KK+1)] = logf(1.0f - red[0]) + norm;
  lm_out[(size_t)bh*(KK+1) + 1 + tid] = lm[s];
}

// ---------------- K4: gather selected K/V rows -------------------------------
__global__ void ann_gather(const float* __restrict__ key_t,
                           const float* __restrict__ value,
                           const int* __restrict__ indices,
                           float* __restrict__ key_out,
                           float* __restrict__ val_out) {
  int bh  = blockIdx.x;
  int grp = blockIdx.y;  // 16 groups of 32 rows
  int tid = threadIdx.x; // 256
  const float4* kb = (const float4*)(key_t + (size_t)bh*SS*DD);
  const float4* vb = (const float4*)(value + (size_t)bh*SS*DD);
  float4* ko = (float4*)(key_out + ((size_t)bh*(KK+1) + 1)*DD);
  float4* vo = (float4*)(val_out + ((size_t)bh*(KK+1) + 1)*DD);
  const int F4 = DD/4; // 32 float4 per row
#pragma unroll
  for (int i = 0; i < 4; ++i) {
    int e   = tid + i*256;       // 0..1023
    int row = grp*32 + e / F4;
    int c   = e % F4;
    int sidx = indices[bh*KK + row];
    ko[(size_t)row*F4 + c] = kb[(size_t)sidx*F4 + c];
    vo[(size_t)row*F4 + c] = vb[(size_t)sidx*F4 + c];
  }
}

// ---------------- K5: finalize mean row, zero row, query copy ----------------
__global__ void ann_final(const float* __restrict__ query,
                          const float* __restrict__ vpartial,
                          float* __restrict__ out_q,
                          float* __restrict__ key_out,
                          float* __restrict__ val_out) {
  int bh = blockIdx.x;
  int d  = threadIdx.x; // 128
  float s = 0.f;
#pragma unroll
  for (int c = 0; c < 8; ++c) s += vpartial[((size_t)bh*8 + c)*DD + d];
  val_out[(size_t)bh*(KK+1)*DD + d] = s / (float)SS;
  key_out[(size_t)bh*(KK+1)*DD + d] = 0.0f;
  out_q  [(size_t)bh*DD + d] = query[(size_t)bh*DD + d];
}

extern "C" void kernel_launch(void* const* d_in, const int* in_sizes, int n_in,
                              void* d_out, int out_size, void* d_ws, size_t ws_size,
                              hipStream_t stream) {
  const float* query   = (const float*)d_in[0];
  const float* key_t   = (const float*)d_in[1];
  const float* value   = (const float*)d_in[2];
  const float* logmask = (const float*)d_in[3];

  float* out     = (float*)d_out;
  float* out_q   = out;
  float* out_key = out_q   + (size_t)BH*DD;
  float* out_val = out_key + (size_t)BH*(KK+1)*DD;
  float* out_lm  = out_val + (size_t)BH*(KK+1)*DD;

  char* ws = (char*)d_ws;
  float* qproj    = (float*)ws; ws += (size_t)BH*RANK_*sizeof(float);
  int*   comp     = (int*)  ws; ws += (size_t)BH*RANK_*sizeof(int);
  float* scores   = (float*)ws; ws += (size_t)BH*SS*sizeof(float);
  float* vpartial = (float*)ws; ws += (size_t)BH*8*DD*sizeof(float);
  int*   indices  = (int*)  ws; ws += (size_t)BH*KK*sizeof(int);

  ann_prep         <<<BH, DD, 0, stream>>>(query, qproj, comp);
  ann_vmean_partial<<<dim3(BH,8), 256, 0, stream>>>(value, vpartial);
  ann_scores       <<<dim3(BH,8), 256, 0, stream>>>(key_t, logmask, qproj, comp, scores);
  size_t sh = (size_t)SS*sizeof(unsigned)            // ub
            + (size_t)CAND*sizeof(unsigned long long) // cand
            + 512*sizeof(float)                       // red
            + 260*sizeof(unsigned);                   // hist + ctl
  ann_topk         <<<BH, 512, sh, stream>>>(scores, logmask, indices, out_lm);
  ann_gather       <<<dim3(BH,16), 256, 0, stream>>>(key_t, value, indices, out_key, out_val);
  ann_final        <<<BH, DD, 0, stream>>>(query, vpartial, out_q, out_key, out_val);
}
